// CircuitGNN_DualTask_12429635355177
// MI455X (gfx1250) — matrix operand order repaired and checked by reading, not compiled, not run on hardware
//
#include <hip/hip_runtime.h>
#include <math.h>

typedef __bf16 bf16;
typedef __attribute__((ext_vector_type(16))) __bf16 v16bf;
typedef __attribute__((ext_vector_type(8)))  __bf16 v8bf;
typedef __attribute__((ext_vector_type(8)))  float  v8f;

#define NEG_SLOPE 0.2f
#define BN_EPS 1e-5f
#define LMAX_SENTINEL 0x007FFFFFu  // encode(-inf)

// ---------- small utility kernels ----------

__global__ void fill_u32_kernel(unsigned* __restrict__ p, unsigned v, long n) {
    long i = (long)blockIdx.x * blockDim.x + threadIdx.x;
    if (i < n) p[i] = v;
}

// X (rows x din) f32 -> dst (rows x kk) bf16, zero-padded in K
__global__ void cvt_x_pad_kernel(const float* __restrict__ src, bf16* __restrict__ dst,
                                 long rows, int din, int kk) {
    long i = (long)blockIdx.x * blockDim.x + threadIdx.x;
    if (i >= rows * kk) return;
    int k = (int)(i % kk);
    long r = i / kk;
    dst[i] = (k < din) ? (bf16)src[r * din + k] : (bf16)0.0f;
}

// W (din x dout) f32 -> Wt (dout x kk) bf16, transposed + zero-padded in K
__global__ void cvt_w_pad_t_kernel(const float* __restrict__ W, bf16* __restrict__ Wt,
                                   int din, int dout, int kk) {
    long i = (long)blockIdx.x * blockDim.x + threadIdx.x;
    if (i >= (long)dout * kk) return;
    int k = (int)(i % kk);
    int c = (int)(i / kk);
    Wt[i] = (k < din) ? (bf16)W[(long)k * dout + c] : (bf16)0.0f;
}

// ---------- ordered-float encoding for atomic max on uint ----------

__device__ __forceinline__ unsigned enc_f32(float f) {
    unsigned u = __float_as_uint(f);
    return (u & 0x80000000u) ? ~u : (u | 0x80000000u);
}
__device__ __forceinline__ float dec_f32(unsigned key) {
    if (key == LMAX_SENTINEL) return 0.0f;  // untouched segment -> where(isfinite,...) => 0
    unsigned u = (key & 0x80000000u) ? (key & 0x7FFFFFFFu) : ~key;
    return __uint_as_float(u);
}

// ---------- WMMA bf16 GEMM:  C[nrows x DOUT] = X[nrows x KK] @ Wt^T + bias ----------
// X is K-padded bf16 (KK multiple of 32); Wt is [DOUT][KK] (transposed, padded).
// Block: ROWT row-tiles of 16 rows; one 16-col tile per wave. B fragments live in
// registers across all row tiles. X tile staged via async-to-LDS DMA (full tiles).
// Fragment layouts: contiguous K order, k = 16*(lane/16) + s, in both A and B.

template <int KK, int DOUT, int ROWT>
__global__ __launch_bounds__(32 * (DOUT / 16))
void gemm_bf16_wmma(const bf16* __restrict__ X, const bf16* __restrict__ Wt,
                    const float* __restrict__ bias, float* __restrict__ C, int nrows) {
    constexpr int KSTEPS = KK / 32;
    constexpr int NW = DOUT / 16;
    constexpr int ROWS = 16 * ROWT;
    __shared__ __align__(64) bf16 xs[ROWS][KK];

    const int row0 = blockIdx.x * ROWS;
    const int lane = threadIdx.x & 31;
    const int wave = threadIdx.x >> 5;
    const int nthr = NW * 32;
    const bool full = (row0 + ROWS) <= nrows;

    if (full) {
        // Linear 16B-chunk DMA into LDS via the CDNA5 async path (ASYNCcnt).
        const bf16* gsrc = X + (size_t)row0 * KK;
        unsigned lds0 = (unsigned)(size_t)(&xs[0][0]);
        constexpr int CHUNKS = ROWS * KK * 2 / 16;
        for (int i = threadIdx.x; i < CHUNKS; i += nthr) {
            unsigned laddr = lds0 + (unsigned)i * 16u;
            unsigned goff = (unsigned)i * 16u;
            asm volatile("global_load_async_to_lds_b128 %0, %1, %2"
                         :: "v"(laddr), "v"(goff), "s"(gsrc) : "memory");
        }
        asm volatile("s_wait_asynccnt 0" ::: "memory");
    } else {
        for (int i = threadIdx.x; i < ROWS * KK; i += nthr) {
            int r = i / KK, k = i % KK;
            int gr = row0 + r;
            xs[r][k] = (gr < nrows) ? X[(size_t)gr * KK + k] : (bf16)0.0f;
        }
    }
    __syncthreads();

    const int col   = wave * 16 + (lane & 15);
    const int akb   = (lane < 16) ? 0 : 16;   // A-fragment K base
    const int bkoff = (lane < 16) ? 0 : 16;   // B-fragment K base

    // B fragments: 16 contiguous bf16 per lane in the transposed/padded weights.
    v16bf bfr[KSTEPS];
#pragma unroll
    for (int s = 0; s < KSTEPS; ++s)
        bfr[s] = *(const v16bf*)(Wt + (size_t)col * KK + s * 32 + bkoff);

    const float bb = bias[col];
    const int mhi = (lane >> 4) * 8;

#pragma unroll
    for (int rt = 0; rt < ROWT; ++rt) {
        const int arow = rt * 16 + (lane & 15);
        v8f acc = {};
#pragma unroll
        for (int s = 0; s < KSTEPS; ++s) {
            const bf16* rowp = &xs[arow][s * 32];
            v8bf lo = *(const v8bf*)(rowp + akb);        // K = akb..akb+7
            v8bf hi = *(const v8bf*)(rowp + 8 + akb);    // K = akb+8..akb+15
            v16bf a = __builtin_shufflevector(lo, hi, 0, 1, 2, 3, 4, 5, 6, 7,
                                              8, 9, 10, 11, 12, 13, 14, 15);
            acc = __builtin_amdgcn_wmma_f32_16x16x32_bf16(
                false, a, false, bfr[s], (short)0, acc, false, false);
        }
        const int mb = row0 + rt * 16 + mhi;
        if (full) {
#pragma unroll
            for (int r = 0; r < 8; ++r)
                C[(size_t)(mb + r) * DOUT + col] = acc[r] + bb;
        } else {
#pragma unroll
            for (int r = 0; r < 8; ++r)
                if (mb + r < nrows) C[(size_t)(mb + r) * DOUT + col] = acc[r] + bb;
        }
    }
}

// ---------- edge pass A: logits + segment max (wave per edge, vector gathers) ----------
// Lanes 0..15 own head 0, lanes 16..31 own head 1; each lane loads EPL contiguous floats.

template <int D>
__global__ void edge_logits_kernel(const float* __restrict__ xl, const float* __restrict__ xr,
                                   const int* __restrict__ ei, const float* __restrict__ att,
                                   float* __restrict__ logits, unsigned* __restrict__ lmaxu,
                                   int E, int E2) {
    constexpr int HD = 2 * D;
    constexpr int EPL = HD / 32;
    typedef float fvec __attribute__((ext_vector_type(EPL)));

    int gw = (int)(((long)blockIdx.x * blockDim.x + threadIdx.x) >> 5);
    int lane = threadIdx.x & 31;
    if (gw >= E2) return;
    int s, d;
    if (gw < E) { s = ei[gw]; d = ei[E + gw]; } else { s = gw - E; d = s; }

    const int c0 = lane * EPL;
    fvec vl = *(const fvec*)(xl + (size_t)s * HD + c0);
    fvec vr = *(const fvec*)(xr + (size_t)d * HD + c0);
    fvec va = *(const fvec*)(att + c0);

    float part = 0.f;
#pragma unroll
    for (int i = 0; i < EPL; ++i) {
        float m = vl[i] + vr[i];
        m = (m > 0.f) ? m : NEG_SLOPE * m;
        part += m * va[i];
    }
    // reduce within each 16-lane head group
#pragma unroll
    for (int off = 1; off <= 8; off <<= 1) part += __shfl_xor(part, off, 32);
    float other = __shfl_xor(part, 16, 32);  // the opposite head's sum

    if (lane == 0) {
        logits[(size_t)gw * 2 + 0] = part;
        logits[(size_t)gw * 2 + 1] = other;
        atomicMax(&lmaxu[(size_t)d * 2 + 0], enc_f32(part));
        atomicMax(&lmaxu[(size_t)d * 2 + 1], enc_f32(other));
    }
}

// ---------- edge pass B: p = exp(l - lmax); denom += p; num += p * xl[src] ----------

template <int D>
__global__ void edge_scatter_kernel(const float* __restrict__ xl, const int* __restrict__ ei,
                                    const float* __restrict__ logits, const unsigned* __restrict__ lmaxu,
                                    float* __restrict__ num, float* __restrict__ den,
                                    int E, int E2) {
    constexpr int HD = 2 * D;
    constexpr int EPL = HD / 32;
    typedef float fvec __attribute__((ext_vector_type(EPL)));

    int gw = (int)(((long)blockIdx.x * blockDim.x + threadIdx.x) >> 5);
    int lane = threadIdx.x & 31;
    if (gw >= E2) return;
    int s, d;
    if (gw < E) { s = ei[gw]; d = ei[E + gw]; } else { s = gw - E; d = s; }

    const int head = lane >> 4;  // lane-uniform over its vector chunk
    float lm = dec_f32(lmaxu[(size_t)d * 2 + head]);
    float p = __expf(logits[(size_t)gw * 2 + head] - lm);

    if ((lane & 15) == 0) atomicAdd(&den[(size_t)d * 2 + head], p);

    const int c0 = lane * EPL;
    fvec vl = *(const fvec*)(xl + (size_t)s * HD + c0);
    float* pn = num + (size_t)d * HD + c0;
#pragma unroll
    for (int i = 0; i < EPL; ++i) atomicAdd(&pn[i], p * vl[i]);
}

// ---------- node combine: y = mean over heads (num/den) + bias; accumulate BN stats ----------

__global__ void node_combine_kernel(const float* __restrict__ num, const float* __restrict__ den,
                                    const float* __restrict__ bias,
                                    float* __restrict__ y, float* __restrict__ sums,
                                    int N, int D) {
    const int HD = 2 * D;
    const int npi = 256 / D;  // nodes per iteration
    const int t = threadIdx.x;
    const int c = t % D;
    const int ln = t / D;
    const int CHUNK = 32;

    float s = 0.f, ss = 0.f;
    long base = (long)blockIdx.x * npi * CHUNK;
    for (int it = 0; it < CHUNK; ++it) {
        long n = base + (long)it * npi + ln;
        if (n < N) {
            float d0 = den[n * 2 + 0] + 1e-16f;
            float d1 = den[n * 2 + 1] + 1e-16f;
            float v = 0.5f * (num[n * HD + c] / d0 + num[n * HD + D + c] / d1) + bias[c];
            y[n * D + c] = v;
            s += v; ss += v * v;
        }
    }
    __shared__ float sb[256], qb[256];
    sb[t] = s; qb[t] = ss;
    __syncthreads();
    for (int str = 128; str >= D; str >>= 1) {
        if (t < str) { sb[t] += sb[t + str]; qb[t] += qb[t + str]; }
        __syncthreads();
    }
    if (t < D) {
        atomicAdd(&sums[c], sb[t]);
        atomicAdd(&sums[D + c], qb[t]);
    }
}

// ---------- BN (training stats, biased var) + ELU + optional residual; emit bf16 copy ----------

__global__ void bn_apply_kernel(const float* __restrict__ y, const float* __restrict__ sums,
                                const float* __restrict__ g, const float* __restrict__ be,
                                float* __restrict__ h, bf16* __restrict__ hbf,
                                int N, int D, int residual) {
    long idx = (long)blockIdx.x * blockDim.x + threadIdx.x;
    if (idx >= (long)N * D) return;
    int c = (int)(idx % D);
    float invN = 1.0f / (float)N;
    float mu = sums[c] * invN;
    float var = sums[D + c] * invN - mu * mu;
    float xn = g[c] * (y[idx] - mu) * rsqrtf(var + BN_EPS) + be[c];
    float z = (xn > 0.f) ? xn : expm1f(xn);
    float hv = residual ? (h[idx] + z) : z;
    h[idx] = hv;
    if (hbf) hbf[idx] = (bf16)hv;
}

// ---------- output heads: imp = h@Wimp + b; pol = sigmoid(h@Wpol + b) (wave per node) ----------

__global__ void heads_kernel(const float* __restrict__ h,
                             const float* __restrict__ Wimp, const float* __restrict__ bimp,
                             const float* __restrict__ Wpol, const float* __restrict__ bpol,
                             float* __restrict__ out, int N) {
    int gw = (int)(((long)blockIdx.x * blockDim.x + threadIdx.x) >> 5);
    int lane = threadIdx.x & 31;
    if (gw >= N) return;
    float hv = h[(long)gw * 32 + lane];
    float a = hv * Wimp[lane];
    float b = hv * Wpol[lane];
#pragma unroll
    for (int off = 16; off; off >>= 1) {
        a += __shfl_xor(a, off, 32);
        b += __shfl_xor(b, off, 32);
    }
    if (lane == 0) {
        out[gw] = a + bimp[0];
        float z = b + bpol[0];
        out[(long)N + gw] = 1.0f / (1.0f + __expf(-z));
    }
}

// ---------- shared edge/node/BN phase (host helper) ----------

template <int D>
static void run_edge_node(float* xl, float* xr_num, const int* ei,
                          const float* att, const float* bias, const float* g, const float* be,
                          float* y, float* h, bf16* hbf_out,
                          unsigned* lmaxu, float* den, float* sums, float* logits,
                          int N, int E, int E2, int residual, hipStream_t stream) {
    const int HD = 2 * D;
    fill_u32_kernel<<<(int)(((long)N * 2 + 255) / 256), 256, 0, stream>>>(lmaxu, LMAX_SENTINEL, (long)N * 2);
    fill_u32_kernel<<<(int)(((long)N * 2 + 255) / 256), 256, 0, stream>>>((unsigned*)den, 0u, (long)N * 2);
    fill_u32_kernel<<<1, 256, 0, stream>>>((unsigned*)sums, 0u, 2 * D);
    edge_logits_kernel<D><<<(E2 + 7) / 8, 256, 0, stream>>>(xl, xr_num, ei, att, logits, lmaxu, E, E2);
    fill_u32_kernel<<<(int)(((long)N * HD + 255) / 256), 256, 0, stream>>>((unsigned*)xr_num, 0u, (long)N * HD);
    edge_scatter_kernel<D><<<(E2 + 7) / 8, 256, 0, stream>>>(xl, ei, logits, lmaxu, xr_num, den, E, E2);
    const int npi = 256 / D;
    const int blocks = (int)(((long)N + (long)npi * 32 - 1) / ((long)npi * 32));
    node_combine_kernel<<<blocks, 256, 0, stream>>>(xr_num, den, bias, y, sums, N, D);
    const long tot = (long)N * D;
    bn_apply_kernel<<<(int)((tot + 255) / 256), 256, 0, stream>>>(y, sums, g, be, h, hbf_out, N, D, residual);
}

// ---------- host orchestration ----------

extern "C" void kernel_launch(void* const* d_in, const int* in_sizes, int n_in,
                              void* d_out, int out_size, void* d_ws, size_t ws_size,
                              hipStream_t stream) {
    const float* x     = (const float*)d_in[0];
    const int*   ei    = (const int*)d_in[1];
    const float* Wl0   = (const float*)d_in[2];
    const float* bl0   = (const float*)d_in[3];
    const float* Wr0   = (const float*)d_in[4];
    const float* br0   = (const float*)d_in[5];
    const float* att0  = (const float*)d_in[6];
    const float* bias0 = (const float*)d_in[7];
    const float* g0    = (const float*)d_in[8];
    const float* be0   = (const float*)d_in[9];
    const float* Wlm   = (const float*)d_in[10];
    const float* blm   = (const float*)d_in[11];
    const float* Wrm   = (const float*)d_in[12];
    const float* brm   = (const float*)d_in[13];
    const float* attm  = (const float*)d_in[14];
    const float* biasm = (const float*)d_in[15];
    const float* gm    = (const float*)d_in[16];
    const float* bem   = (const float*)d_in[17];
    const float* WlL   = (const float*)d_in[18];
    const float* blL   = (const float*)d_in[19];
    const float* WrL   = (const float*)d_in[20];
    const float* brL   = (const float*)d_in[21];
    const float* attL  = (const float*)d_in[22];
    const float* biasL = (const float*)d_in[23];
    const float* gL    = (const float*)d_in[24];
    const float* beL   = (const float*)d_in[25];
    const float* Wimp  = (const float*)d_in[26];
    const float* bimp  = (const float*)d_in[27];
    const float* Wpol  = (const float*)d_in[28];
    const float* bpol  = (const float*)d_in[29];

    const int F = 16, HID = 64, OUT = 32, N_MID = 18;
    const int N  = in_sizes[0] / F;
    const int E  = in_sizes[1] / 2;
    const int E2 = E + N;
    (void)n_in; (void)out_size; (void)ws_size;

    // workspace layout
    char* ws = (char*)d_ws;
    size_t off = 0;
    auto walloc = [&](size_t bytes) -> void* {
        void* p = ws + off;
        off += (bytes + 255) & ~(size_t)255;
        return p;
    };
    float*    xl     = (float*)walloc((size_t)N * 128 * 4);
    float*    xr_num = (float*)walloc((size_t)N * 128 * 4);  // xr during pass A, then num
    float*    h      = (float*)walloc((size_t)N * 64 * 4);
    float*    y      = (float*)walloc((size_t)N * 64 * 4);
    bf16*     hbf    = (bf16*)walloc((size_t)N * 64 * 2);
    bf16*     xbf    = (bf16*)walloc((size_t)N * 32 * 2);  // layer-0 X, K-padded to 32
    float*    logits = (float*)walloc((size_t)E2 * 2 * 4);
    unsigned* lmaxu  = (unsigned*)walloc((size_t)N * 2 * 4);
    float*    den    = (float*)walloc((size_t)N * 2 * 4);
    float*    sums   = (float*)walloc(128 * 4);
    bf16*     wbl    = (bf16*)walloc(128 * 64 * 2);  // Wt: [DOUT][KK] bf16
    bf16*     wbr    = (bf16*)walloc(128 * 64 * 2);

    constexpr int ROWT = 5;                       // 50000 = 625 * 80 -> exact tiling
    const int gblocks = (N + 16 * ROWT - 1) / (16 * ROWT);

    auto cvtw = [&](const float* W, bf16* Wt, int din, int dout, int kk) {
        long n = (long)dout * kk;
        cvt_w_pad_t_kernel<<<(int)((n + 255) / 256), 256, 0, stream>>>(W, Wt, din, dout, kk);
    };

    // ----- layer 0: F(16) -> HID(64), heads 2 (KK = 32, DOUT = 128) -----
    {
        long n = (long)N * 32;
        cvt_x_pad_kernel<<<(int)((n + 255) / 256), 256, 0, stream>>>(x, xbf, N, F, 32);
    }
    cvtw(Wl0, wbl, F, 128, 32);
    cvtw(Wr0, wbr, F, 128, 32);
    gemm_bf16_wmma<32, 128, ROWT><<<gblocks, 256, 0, stream>>>(xbf, wbl, bl0, xl, N);
    gemm_bf16_wmma<32, 128, ROWT><<<gblocks, 256, 0, stream>>>(xbf, wbr, br0, xr_num, N);
    run_edge_node<64>(xl, xr_num, ei, att0, bias0, g0, be0, y, h, hbf,
                      lmaxu, den, sums, logits, N, E, E2, 0, stream);

    // ----- 18 middle layers: 64 -> 64, residual -----
    for (int l = 0; l < N_MID; ++l) {
        cvtw(Wlm + (long)l * 64 * 128, wbl, 64, 128, 64);
        cvtw(Wrm + (long)l * 64 * 128, wbr, 64, 128, 64);
        gemm_bf16_wmma<64, 128, ROWT><<<gblocks, 256, 0, stream>>>(hbf, wbl, blm + (long)l * 128, xl, N);
        gemm_bf16_wmma<64, 128, ROWT><<<gblocks, 256, 0, stream>>>(hbf, wbr, brm + (long)l * 128, xr_num, N);
        run_edge_node<64>(xl, xr_num, ei, attm + (long)l * 128, biasm + (long)l * 64,
                          gm + (long)l * 64, bem + (long)l * 64, y, h, hbf,
                          lmaxu, den, sums, logits, N, E, E2, 1, stream);
    }

    // ----- last layer: 64 -> 32, heads 2 (KK = 64, DOUT = 64) -----
    cvtw(WlL, wbl, 64, 64, 64);
    cvtw(WrL, wbr, 64, 64, 64);
    gemm_bf16_wmma<64, 64, ROWT><<<gblocks, 128, 0, stream>>>(hbf, wbl, blL, xl, N);
    gemm_bf16_wmma<64, 64, ROWT><<<gblocks, 128, 0, stream>>>(hbf, wbr, brL, xr_num, N);
    run_edge_node<32>(xl, xr_num, ei, attL, biasL, gL, beL, y, h, nullptr,
                      lmaxu, den, sums, logits, N, E, E2, 0, stream);

    // ----- output heads -----
    heads_kernel<<<(N + 7) / 8, 256, 0, stream>>>(h, Wimp, bimp, Wpol, bpol, (float*)d_out, N);
}
